// VisionAttention_77841987273029
// MI455X (gfx1250) — matrix operand order repaired and checked by reading, not compiled, not run on hardware
//
#include <hip/hip_runtime.h>

// ---------------------------------------------------------------------------
// VisionAttention for MI455X (gfx1250, wave32, WMMA bf16 16x16x32, f32 accum)
// S=8192 tokens, DIM=1280, H=16 heads, HD=80 (padded to 96), B=8, L=1024
// ---------------------------------------------------------------------------

#define S_TOK 8192
#define DIMM  1280
#define NH    16
#define HD    80
#define HDP   96          // padded head dim (3 x 32 K-steps)
#define SEG   8
#define SEQ   1024

typedef __bf16 v16bf __attribute__((ext_vector_type(16)));
typedef float  v8f   __attribute__((ext_vector_type(8)));
typedef unsigned int u32x4 __attribute__((ext_vector_type(4)));

// Load 16 bf16 (two 16-byte chunks) into one WMMA operand register set.
__device__ inline v16bf ld2x16(const __bf16* p0, const __bf16* p1) {
    union U { u32x4 u[2]; v16bf v; } x;
    x.u[0] = *reinterpret_cast<const u32x4*>(p0);
    x.u[1] = *reinterpret_cast<const u32x4*>(p1);
    return x.v;
}

__device__ inline v8f wmma_bf16(v16bf a, v16bf b, v8f c) {
    return __builtin_amdgcn_wmma_f32_16x16x32_bf16(false, a, false, b,
                                                   (short)0, c, false, false);
}

// xor-16 lane exchange via ds_swizzle (SWAPX16): and=0x1f, or=0, xor=0x10
__device__ inline float xor16(float x) {
    int i = __builtin_amdgcn_ds_swizzle(__float_as_int(x), 0x401f);
    return __int_as_float(i);
}

// ---------------------------------------------------------------------------
// Elementwise fp32 -> bf16
__global__ void va_cvt_bf16(const float* __restrict__ src,
                            __bf16* __restrict__ dst, long n) {
    long i = (long)blockIdx.x * blockDim.x + threadIdx.x;
    if (i < n) dst[i] = (__bf16)src[i];
}

// src[K][N] fp32 (row-major)  ->  dst[N][K] bf16  (weights pre-transposed so
// the WMMA B operand reads contiguous K per lane)
__global__ void va_transpose_bf16(const float* __restrict__ src,
                                  __bf16* __restrict__ dst, int K, int N) {
    long i = (long)blockIdx.x * blockDim.x + threadIdx.x;
    if (i >= (long)K * N) return;
    int k = (int)(i % K);
    int n = (int)(i / K);
    dst[i] = (__bf16)src[(long)k * N + n];
}

// ---------------------------------------------------------------------------
// Tiled bf16 WMMA GEMM:  C[M,N] = A[M,K] * BT[N,K]^T + bias[N]
// One wave computes a 32x64 tile (2 M-subtiles x 4 N-subtiles).
__global__ __launch_bounds__(256)
void va_gemm_bf16(const __bf16* __restrict__ A, const __bf16* __restrict__ BT,
                  const float* __restrict__ bias,
                  float* __restrict__ outF, __bf16* __restrict__ outB,
                  int M, int N, int K) {
    int wid  = (int)((blockIdx.x * blockDim.x + threadIdx.x) >> 5);
    int nt64 = N >> 6;
    int total = (M >> 5) * nt64;
    if (wid >= total) return;
    int mt = wid / nt64;
    int nt = wid - mt * nt64;
    int lane = threadIdx.x & 31;
    int n16  = lane & 15;
    int kh   = lane >> 4;

    v8f acc[2][4];
#pragma unroll
    for (int mi = 0; mi < 2; ++mi)
#pragma unroll
        for (int s = 0; s < 4; ++s)
#pragma unroll
            for (int r = 0; r < 8; ++r) acc[mi][s][r] = 0.f;

    const __bf16* a0p = A + (long)(mt * 32 + n16) * K + kh * 16;
    const __bf16* a1p = a0p + (long)16 * K;
    const __bf16* bp[4];
#pragma unroll
    for (int s = 0; s < 4; ++s)
        bp[s] = BT + (long)(nt * 64 + s * 16 + n16) * K + kh * 16;

    for (int kb = 0; kb < K; kb += 32) {
        v16bf a0 = ld2x16(a0p + kb, a0p + kb + 8);
        v16bf a1 = ld2x16(a1p + kb, a1p + kb + 8);
#pragma unroll
        for (int s = 0; s < 4; ++s) {
            v16bf bb = ld2x16(bp[s] + kb, bp[s] + kb + 8);
            acc[0][s] = wmma_bf16(a0, bb, acc[0][s]);
            acc[1][s] = wmma_bf16(a1, bb, acc[1][s]);
        }
    }

#pragma unroll
    for (int s = 0; s < 4; ++s) {
        int col = nt * 64 + s * 16 + n16;
        float bv = bias[col];
#pragma unroll
        for (int mi = 0; mi < 2; ++mi) {
#pragma unroll
            for (int r = 0; r < 8; ++r) {
                long row = mt * 32 + mi * 16 + kh * 8 + r;
                float v = acc[mi][s][r] + bv;
                if (outB) outB[row * N + col] = (__bf16)v;
                else      outF[row * N + col] = v;
            }
        }
    }
}

// ---------------------------------------------------------------------------
// RoPE + relayout: qkv[S][3*DIM] bf16 ->
//   qarr [b,h,t,HDP] bf16 (q*cos + rot(q)*sin, pre-scaled by HD^-0.5, padded)
//   karr [b,h,t,HDP] bf16
//   vt   [b,h,hd,t]  bf16 (V transposed -> A operand of PV GEMM)
__global__ void va_rope(const __bf16* __restrict__ qkv,
                        const float* __restrict__ cosA,
                        const float* __restrict__ sinA,
                        __bf16* __restrict__ qarr, __bf16* __restrict__ karr,
                        __bf16* __restrict__ vt) {
    long idx = (long)blockIdx.x * blockDim.x + threadIdx.x;
    if (idx >= (long)S_TOK * NH * HDP) return;
    int hdp = (int)(idx % HDP);
    long th = idx / HDP;
    int h = (int)(th % NH);
    int t = (int)(th / NH);
    int b  = t >> 10;
    int tt = t & 1023;
    long qdst = ((long)(b * NH + h) * SEQ + tt) * HDP + hdp;
    if (hdp >= HD) { qarr[qdst] = (__bf16)0.f; karr[qdst] = (__bf16)0.f; return; }
    int hd = hdp;
    const float scale = 0.11180339887498949f;   // HD^-0.5
    long base = (long)t * (3 * DIMM) + h * HD;
    float qv = (float)qkv[base + hd];
    float kv = (float)qkv[base + DIMM + hd];
    float vv = (float)qkv[base + 2 * DIMM + hd];
    int ph = (hd < HD / 2) ? hd + HD / 2 : hd - HD / 2;
    float sgn = (hd < HD / 2) ? -1.f : 1.f;
    float qr = sgn * (float)qkv[base + ph];
    float kr = sgn * (float)qkv[base + DIMM + ph];
    float c = cosA[(long)t * HD + hd];
    float s = sinA[(long)t * HD + hd];
    qarr[qdst] = (__bf16)((qv * c + qr * s) * scale);
    karr[qdst] = (__bf16)(kv * c + kr * s);
    vt[((long)(b * NH + h) * HD + hd) * SEQ + tt] = (__bf16)vv;
}

// ---------------------------------------------------------------------------
// Flash attention (transposed form). One wave owns 16 queries of one (b,h).
//   S^T = K_mat * Q^T   (A = K rows, B = Q; query pinned to lane N)
//   O^T = V^T  * P^T    (A = V^T rows, B = P^T built with xor-16 shuffles)
__global__ __launch_bounds__(128)
void va_attn(const __bf16* __restrict__ qarr, const __bf16* __restrict__ karr,
             const __bf16* __restrict__ vt, __bf16* __restrict__ aout) {
    int wid  = (int)((blockIdx.x * blockDim.x + threadIdx.x) >> 5);
    int lane = threadIdx.x & 31;
    int qt = wid & 63;           // 64 query tiles of 16
    int bh = wid >> 6;           // b*16 + h
    int h = bh & 15;
    int b = bh >> 4;
    int n  = lane & 15;          // query within tile / matrix row within tile
    int kh = lane >> 4;

    const __bf16* qb = qarr + ((long)bh * SEQ + qt * 16) * HDP;
    const __bf16* kb = karr + (long)bh * SEQ * HDP;
    const __bf16* vb = vt   + (long)bh * HD * SEQ;

    // Q as B operand: lane = query, contiguous head-dim; 3 K-steps of 32.
    v16bf Bq[3];
#pragma unroll
    for (int t = 0; t < 3; ++t) {
        const __bf16* p = qb + (long)n * HDP + t * 32 + kh * 16;
        Bq[t] = ld2x16(p, p + 8);
    }

    v8f o[5];
#pragma unroll
    for (int d = 0; d < 5; ++d)
#pragma unroll
        for (int r = 0; r < 8; ++r) o[d][r] = 0.f;
    float mi = -1e30f, li = 0.f;

    for (int kc = 0; kc < SEQ; kc += 32) {
        // S^T for 32 keys: two 16-key C tiles
        v8f s0, s1;
#pragma unroll
        for (int r = 0; r < 8; ++r) { s0[r] = 0.f; s1[r] = 0.f; }
#pragma unroll
        for (int t = 0; t < 3; ++t) {
            const __bf16* p0 = kb + (long)(kc + n) * HDP + t * 32 + kh * 16;
            const __bf16* p1 = kb + (long)(kc + 16 + n) * HDP + t * 32 + kh * 16;
            v16bf ak0 = ld2x16(p0, p0 + 8);
            v16bf ak1 = ld2x16(p1, p1 + 8);
            s0 = wmma_bf16(ak0, Bq[t], s0);
            s1 = wmma_bf16(ak1, Bq[t], s1);
        }
        // online softmax (per-query state lives per lane; halves synced by xor16)
        float cm = -1e30f;
#pragma unroll
        for (int r = 0; r < 8; ++r) {
            cm = fmaxf(cm, s0[r]); cm = fmaxf(cm, s1[r]);
        }
        cm = fmaxf(cm, xor16(cm));
        float nm = fmaxf(mi, cm);
        float p0v[8], p1v[8];
        float rs = 0.f;
#pragma unroll
        for (int r = 0; r < 8; ++r) {
            p0v[r] = __expf(s0[r] - nm);
            p1v[r] = __expf(s1[r] - nm);
            rs += p0v[r] + p1v[r];
        }
        rs += xor16(rs);
        float alpha = __expf(mi - nm);
        li = li * alpha + rs;
        mi = nm;
#pragma unroll
        for (int d = 0; d < 5; ++d)
#pragma unroll
            for (int r = 0; r < 8; ++r) o[d][r] *= alpha;

        // Build P^T B-operand (lane = query, elems = 16 keys of this half)
        v16bf pt;
#pragma unroll
        for (int r = 0; r < 8; ++r) {
            float a0 = xor16(p0v[r]);
            float a1 = xor16(p1v[r]);
            float lo = (kh == 0) ? p0v[r] : a1;   // keys kh*16 + r
            float hi = (kh == 0) ? a0 : p1v[r];   // keys kh*16 + 8 + r
            pt[r]     = (__bf16)lo;
            pt[8 + r] = (__bf16)hi;
        }
        // O^T += V^T * P^T over 5 head-dim tiles
#pragma unroll
        for (int d = 0; d < 5; ++d) {
            const __bf16* p = vb + (long)(d * 16 + n) * SEQ + kc + kh * 16;
            v16bf av = ld2x16(p, p + 8);
            o[d] = wmma_bf16(av, pt, o[d]);
        }
    }

    float inv = 1.f / li;
    long token = (long)b * SEQ + qt * 16 + n;
    __bf16* op = aout + token * DIMM + h * HD;
#pragma unroll
    for (int d = 0; d < 5; ++d)
#pragma unroll
        for (int r = 0; r < 8; ++r)
            op[d * 16 + kh * 8 + r] = (__bf16)(o[d][r] * inv);
}

// ---------------------------------------------------------------------------
extern "C" void kernel_launch(void* const* d_in, const int* in_sizes, int n_in,
                              void* d_out, int out_size, void* d_ws, size_t ws_size,
                              hipStream_t stream) {
    const float* hidden = (const float*)d_in[0];
    const float* cosA   = (const float*)d_in[1];
    const float* sinA   = (const float*)d_in[2];
    const float* Wqkv   = (const float*)d_in[3];
    const float* bqkv   = (const float*)d_in[4];
    const float* Wproj  = (const float*)d_in[5];
    const float* bproj  = (const float*)d_in[6];
    float* out = (float*)d_out;

    char* ws = (char*)d_ws;
    size_t off = 0;
    __bf16* hb   = (__bf16*)(ws + off); off += (size_t)S_TOK * DIMM * 2;        // 20.97 MB
    __bf16* wqT  = (__bf16*)(ws + off); off += (size_t)3 * DIMM * DIMM * 2;     //  9.83 MB
    __bf16* wpT  = (__bf16*)(ws + off); off += (size_t)DIMM * DIMM * 2;         //  3.28 MB
    __bf16* qkvb = (__bf16*)(ws + off); off += (size_t)S_TOK * 3 * DIMM * 2;    // 62.91 MB
    __bf16* qarr = (__bf16*)(ws + off); off += (size_t)S_TOK * NH * HDP * 2 / NH * NH; // 25.17 MB
    __bf16* karr = (__bf16*)(ws + off); off += (size_t)S_TOK * NH * HDP * 2 / NH * NH; // 25.17 MB
    __bf16* vtr  = (__bf16*)(ws + off); off += (size_t)S_TOK * DIMM * 2;        // 20.97 MB
    __bf16* aout = (__bf16*)(ws + off); off += (size_t)S_TOK * DIMM * 2;        // 20.97 MB
    (void)ws_size; (void)in_sizes; (void)n_in; (void)out_size;

    // 1) convert inputs to bf16 (weights transposed for B-operand layout)
    {
        long n = (long)S_TOK * DIMM;
        va_cvt_bf16<<<(unsigned)((n + 255) / 256), 256, 0, stream>>>(hidden, hb, n);
    }
    {
        long n = (long)DIMM * 3 * DIMM;
        va_transpose_bf16<<<(unsigned)((n + 255) / 256), 256, 0, stream>>>(Wqkv, wqT, DIMM, 3 * DIMM);
    }
    {
        long n = (long)DIMM * DIMM;
        va_transpose_bf16<<<(unsigned)((n + 255) / 256), 256, 0, stream>>>(Wproj, wpT, DIMM, DIMM);
    }

    // 2) QKV GEMM: [8192,1280] x [1280,3840] + bias -> bf16
    {
        int waves = (S_TOK / 32) * (3 * DIMM / 64);   // 15360
        va_gemm_bf16<<<(waves * 32 + 255) / 256, 256, 0, stream>>>(
            hb, wqT, bqkv, nullptr, qkvb, S_TOK, 3 * DIMM, DIMM);
    }

    // 3) RoPE + relayout (q scaled, HD padded to 96, V transposed)
    {
        long n = (long)S_TOK * NH * HDP;
        va_rope<<<(unsigned)((n + 255) / 256), 256, 0, stream>>>(
            qkvb, cosA, sinA, qarr, karr, vtr);
    }

    // 4) Flash attention: 8192 waves (b,h,qtile), 4 waves / block
    va_attn<<<(SEG * NH * (SEQ / 16)) / 4, 128, 0, stream>>>(qarr, karr, vtr, aout);

    // 5) Output projection -> fp32 d_out
    {
        int waves = (S_TOK / 32) * (DIMM / 64);       // 5120
        va_gemm_bf16<<<(waves * 32 + 255) / 256, 256, 0, stream>>>(
            aout, wpT, bproj, out, nullptr, S_TOK, DIMM, DIMM);
    }
}
